// RVQHead_21586505629895
// MI455X (gfx1250) — matrix operand order repaired and checked by reading, not compiled, not run on hardware
//
#include <hip/hip_runtime.h>
#include <hip/hip_bf16.h>

// ---------------------------------------------------------------------------
// RVQ head for MI455X (gfx1250, wave32).  Matmuls via v_wmma_f32_16x16x32_bf16,
// LDS staging via CDNA5 async copy engine (global_load_async_to_lds_b128),
// double-buffered software pipeline (1 barrier / K-step).
// ---------------------------------------------------------------------------

typedef __attribute__((ext_vector_type(16))) __bf16 v16bf;
typedef __attribute__((ext_vector_type(8)))  __bf16 v8bf;
typedef __attribute__((ext_vector_type(8)))  float  v8f;

union FragAB { v16bf v; v8bf h[2]; };

static __device__ __forceinline__ unsigned short f2bf(float f) {
    unsigned int u = __builtin_bit_cast(unsigned int, f);
    u = (u + 0x7FFFu + ((u >> 16) & 1u)) >> 16;     // round-to-nearest-even
    return (unsigned short)u;
}

// 16B-per-lane async DMA global -> LDS (ASYNCcnt tracked).  ldsPtr's low 32
// bits are the LDS byte offset (generic LDS pointers carry it in addr[31:0]).
static __device__ __forceinline__ void async_ld16(void* ldsPtr, const void* gptr) {
    unsigned lds = (unsigned)(unsigned long long)ldsPtr;
    asm volatile("global_load_async_to_lds_b128 %0, %1, off"
                 :: "v"(lds), "v"(gptr) : "memory");
}
static __device__ __forceinline__ void wait_async() {
    asm volatile("s_wait_asynccnt 0x0" ::: "memory");
}

#define N_ROWS 32768          // B*T
#define DIM    512            // embed dim
#define INCH   1024
#define NCODE  1024
#define NLVL   4

// ---------------------------------------------------------------------------
// Generic bf16 WMMA GEMM:  C[M x Nn] = A[M x K] * BT[Nn x K]^T  (+bias, +relu)
// Block tile 64x128, 8 waves (4 along M x 2 along N), wave tile 16x64.
// Double-buffered LDS tiles, async-DMA staged, 1 barrier per K-step.
// MODE 0: bias, write f32 to Cf and Cf2, bf16 to Cb      (projection)
// MODE 1: bias+relu, write bf16 to Cb                    (decoder hidden)
// MODE 2: bias, write f32 to Cf                          (decoder out)
// ---------------------------------------------------------------------------
template <int MODE>
__global__ __launch_bounds__(256)
void gemm_bf16(const unsigned short* __restrict__ A,
               const unsigned short* __restrict__ BT,
               const float* __restrict__ bias,
               float* __restrict__ Cf, float* __restrict__ Cf2,
               unsigned short* __restrict__ Cb,
               int K, int ldc)
{
    constexpr int BM = 64, BN = 128, BK = 32, LST = 40; // LDS k-stride (pad)
    __shared__ __align__(16) unsigned short As[2][BM][LST];
    __shared__ __align__(16) unsigned short Bs[2][BN][LST];

    const int tid  = threadIdx.x;
    const int lane = tid & 31;
    const int wave = tid >> 5;
    const int wm   = wave & 3;       // M sub-tile (x16)
    const int wn   = wave >> 2;      // N sub-tile (x64)
    const int half = lane >> 4;
    const int l15  = lane & 15;
    const long mBase = (long)blockIdx.y * BM;
    const long nBase = (long)blockIdx.x * BN;

    // per-thread staging slots (one 16B A chunk, two 16B B chunks)
    const int arow = tid >> 2, acg = (tid & 3) << 3;
    const int brow = tid >> 1, bcg = (tid & 1) << 4;
    const unsigned short* aSrc = A + (mBase + arow) * (long)K + acg;
    const unsigned short* bSrc = BT + (nBase + brow) * (long)K + bcg;

    v8f acc[4] = {};
    const int nsteps = K / BK;

    // prime buffer 0
    async_ld16(&As[0][arow][acg], aSrc);
    async_ld16(&Bs[0][brow][bcg], bSrc);
    async_ld16(&Bs[0][brow][bcg + 8], bSrc + 8);

    for (int s = 0; s < nsteps; ++s) {
        const int cur = s & 1;
        wait_async();          // my async writes into buf `cur` have landed
        __syncthreads();       // everyone's have; everyone done reading buf 1-cur
        if (s + 1 < nsteps) {  // prefetch next tile into the other buffer
            const int k1 = (s + 1) * BK;
            async_ld16(&As[1 - cur][arow][acg], aSrc + k1);
            async_ld16(&Bs[1 - cur][brow][bcg], bSrc + k1);
            async_ld16(&Bs[1 - cur][brow][bcg + 8], bSrc + k1 + 8);
        }

        FragAB af;  // A 16x32 fragment: lane half selects K {0..15} vs {16..31}
        {
            int row = wm * 16 + l15;
            af.h[0] = *(const v8bf*)&As[cur][row][half * 16];
            af.h[1] = *(const v8bf*)&As[cur][row][half * 16 + 8];
        }
        FragAB bf[4]; // preload all B fragments -> batch ds_loads under one wait
#pragma unroll
        for (int t = 0; t < 4; ++t) {
            int col = wn * 64 + t * 16 + l15;
            bf[t].h[0] = *(const v8bf*)&Bs[cur][col][half * 16];
            bf[t].h[1] = *(const v8bf*)&Bs[cur][col][half * 16 + 8];
        }
#pragma unroll
        for (int t = 0; t < 4; ++t)
            acc[t] = __builtin_amdgcn_wmma_f32_16x16x32_bf16(
                false, af.v, false, bf[t].v, (short)0, acc[t], false, false);
    }

#pragma unroll
    for (int t = 0; t < 4; ++t) {
        long col = nBase + wn * 64 + t * 16 + l15;
        float bv = bias[col];
#pragma unroll
        for (int r = 0; r < 8; ++r) {
            long row = mBase + wm * 16 + r + half * 8;
            float v = acc[t][r] + bv;
            if (MODE == 1) v = fmaxf(v, 0.0f);
            long off = row * (long)ldc + col;
            if (MODE == 0) { Cf[off] = v; Cf2[off] = v; Cb[off] = f2bf(v); }
            else if (MODE == 1) { Cb[off] = f2bf(v); }
            else { Cf[off] = v; }
        }
    }
}

// ---------------------------------------------------------------------------
// Fused RVQ level: scores = ||c||^2 - 2 r.c over all 1024 codes, argmin per row.
// 32 rows/block cached fully in LDS (74.5KB total, exploiting 320KB WGP LDS);
// 8 waves = 2Mx4N, wave tile 16x64; codebook tiles double-buffered via async
// DMA, 1 barrier per K-step; block argmin via ds_min_u64 on packed (score||idx).
// ---------------------------------------------------------------------------
__global__ __launch_bounds__(256)
void rvq_score_kernel(const unsigned short* __restrict__ Rb,   // [N x 512] residual bf16
                      const unsigned short* __restrict__ CB,   // [1024 x 512] codebook bf16
                      const float* __restrict__ cnorm,         // [1024]
                      int* __restrict__ indices,
                      unsigned int* __restrict__ counts,
                      int level)
{
    constexpr int SBM = 32, SBN = 256, AST = DIM + 8, LST = 40;
    __shared__ __align__(16) unsigned short As[SBM][AST];      // 32 x 520 bf16
    __shared__ __align__(16) unsigned short Bs[2][SBN][LST];   // 2 x 256 x 40 bf16
    __shared__ unsigned long long best[SBM];

    const int tid  = threadIdx.x;
    const int lane = tid & 31;
    const int wave = tid >> 5;
    const int wm   = wave & 1;       // M sub-tile (x16)
    const int wn   = wave >> 1;      // N sub-tile (x64)
    const int half = lane >> 4;
    const int l15  = lane & 15;
    const long mBase = (long)blockIdx.x * SBM;

    if (tid < SBM) best[tid] = ~0ull;

    // B staging: 4 chunks of 8 bf16 per thread; flat step st = chunk*16 + kstep
    auto stageB = [&](int buf, int st) {
        const int nb = (st >> 4) << 8;          // 256-code chunk base
        const int k0 = (st & 15) << 5;          // K offset
#pragma unroll
        for (int i = 0; i < 4; ++i) {
            int idx = tid + i * 256;
            int row = idx >> 2, cg = (idx & 3) << 3;
            async_ld16(&Bs[buf][row][cg], CB + (long)(nb + row) * DIM + k0 + cg);
        }
    };

    // stage all 32 residual rows once + prime B buffer 0 (one wait covers both)
#pragma unroll
    for (int i = 0; i < 8; ++i) {
        int idx = tid + i * 256;          // chunk of 8 bf16; 64 chunks/row
        int row = idx >> 6, cg = (idx & 63) << 3;
        async_ld16(&As[row][cg], Rb + (mBase + row) * (long)DIM + cg);
    }
    stageB(0, 0);

    for (int chunk = 0; chunk < 4; ++chunk) {
        v8f acc[4] = {};
        for (int ks = 0; ks < 16; ++ks) {
            const int st  = chunk * 16 + ks;
            const int cur = st & 1;
            wait_async();
            __syncthreads();
            if (st + 1 < 64) stageB(1 - cur, st + 1);

            const int k0 = ks << 5;
            FragAB af;
            {
                int row = wm * 16 + l15;
                af.h[0] = *(const v8bf*)&As[row][k0 + half * 16];
                af.h[1] = *(const v8bf*)&As[row][k0 + half * 16 + 8];
            }
            FragAB bf[4];
#pragma unroll
            for (int t = 0; t < 4; ++t) {
                int col = wn * 64 + t * 16 + l15;
                bf[t].h[0] = *(const v8bf*)&Bs[cur][col][half * 16];
                bf[t].h[1] = *(const v8bf*)&Bs[cur][col][half * 16 + 8];
            }
#pragma unroll
            for (int t = 0; t < 4; ++t)
                acc[t] = __builtin_amdgcn_wmma_f32_16x16x32_bf16(
                    false, af.v, false, bf[t].v, (short)0, acc[t], false, false);
        }
        // per-row argmin over this 256-code chunk
        const int nb = chunk << 8;
#pragma unroll
        for (int r = 0; r < 8; ++r) {
            float bs = 3.4e38f; int bj = 0;
#pragma unroll
            for (int t = 0; t < 4; ++t) {
                int j = nb + wn * 64 + t * 16 + l15;
                float s = cnorm[j] - 2.0f * acc[t][r];
                if (s < bs) { bs = s; bj = j; }
            }
            // reduce across the 16 lanes holding this row
#pragma unroll
            for (int off = 8; off >= 1; off >>= 1) {
                float os = __shfl_xor(bs, off, 32);
                int   oj = __shfl_xor(bj, off, 32);
                if (os < bs || (os == bs && oj < bj)) { bs = os; bj = oj; }
            }
            if (l15 == 0) {
                unsigned int u = __builtin_bit_cast(unsigned int, bs);
                u = ((int)u < 0) ? ~u : (u | 0x80000000u);   // order-preserving map
                unsigned long long pk = ((unsigned long long)u << 32) | (unsigned)bj;
                atomicMin(&best[wm * 16 + r + half * 8], pk); // -> ds_min_u64
            }
        }
    }
    __syncthreads();
    if (tid < SBM) {
        int idx = (int)(best[tid] & 0xFFFFFFFFu);
        long n = mBase + tid;
        indices[n * NLVL + level] = idx;
        atomicAdd(&counts[idx], 1u);
    }
}

// residual update: r -= cb[idx]; refresh bf16 copy (one float4 per thread)
__global__ void rvq_gather_kernel(const float* __restrict__ cbf,
                                  const int* __restrict__ indices,
                                  float* __restrict__ residf,
                                  unsigned short* __restrict__ residb,
                                  int level)
{
    long i = (long)blockIdx.x * blockDim.x + threadIdx.x;   // N*128 threads
    long n = i >> 7;
    int  d = (int)(i & 127) << 2;
    int  code = indices[n * NLVL + level];
    const float4 c = *(const float4*)(cbf + ((long)level * NCODE + code) * DIM + d);
    float4 r = *(float4*)(residf + n * DIM + d);
    r.x -= c.x; r.y -= c.y; r.z -= c.z; r.w -= c.w;
    *(float4*)(residf + n * DIM + d) = r;
    unsigned long long pk = (unsigned long long)f2bf(r.x)
                          | ((unsigned long long)f2bf(r.y) << 16)
                          | ((unsigned long long)f2bf(r.z) << 32)
                          | ((unsigned long long)f2bf(r.w) << 48);
    *(unsigned long long*)(residb + n * DIM + d) = pk;
}

// q_total = proj - resid (bf16 for decoder); accumulate sum(resid^2) for losses
__global__ void rvq_final_kernel(const float* __restrict__ proj,
                                 const float* __restrict__ residf,
                                 unsigned short* __restrict__ qtotb,
                                 float* __restrict__ lossAcc)
{
    long i = (long)blockIdx.x * blockDim.x + threadIdx.x;   // N*512/4 threads
    long o = i << 2;
    float4 p = *(const float4*)(proj + o);
    float4 r = *(const float4*)(residf + o);
    float4 q = {p.x - r.x, p.y - r.y, p.z - r.z, p.w - r.w};
    unsigned long long pk = (unsigned long long)f2bf(q.x)
                          | ((unsigned long long)f2bf(q.y) << 16)
                          | ((unsigned long long)f2bf(q.z) << 32)
                          | ((unsigned long long)f2bf(q.w) << 48);
    *(unsigned long long*)(qtotb + o) = pk;
    float s = r.x * r.x + r.y * r.y + r.z * r.z + r.w * r.w;
#pragma unroll
    for (int off = 16; off >= 1; off >>= 1) s += __shfl_xor(s, off, 32);
    if ((threadIdx.x & 31) == 0) atomicAdd(lossAcc, s);
}

// ---- conversion / layout helpers -----------------------------------------
__global__ void init_kernel(unsigned int* counts, float* lossAcc) {
    counts[threadIdx.x] = 0u;
    if (threadIdx.x == 0) lossAcc[0] = 0.0f;
}
__global__ void convert_transpose_kernel(const float* __restrict__ in,
                                         unsigned short* __restrict__ out,
                                         int rows, int cols, int colsPad)
{   // out[j][i] = (j<cols) ? in[i][j] : 0   ; out is [colsPad x rows]
    long i = (long)blockIdx.x * blockDim.x + threadIdx.x;
    if (i >= (long)colsPad * rows) return;
    int r = (int)(i % rows), j = (int)(i / rows);
    float v = (j < cols) ? in[(long)r * cols + j] : 0.0f;
    out[i] = f2bf(v);
}
__global__ void cvt_kernel(const float* __restrict__ in,
                           unsigned short* __restrict__ out, long n) {
    long i = (long)blockIdx.x * blockDim.x + threadIdx.x;
    if (i < n) out[i] = f2bf(in[i]);
}
__global__ void cbnorm_kernel(const float* __restrict__ cb, float* __restrict__ nrm) {
    int i = blockIdx.x * blockDim.x + threadIdx.x;   // 4096 codes total
    const float* p = cb + (long)i * DIM;
    float s = 0.0f;
    for (int d = 0; d < DIM; d += 4) {
        float4 v = *(const float4*)(p + d);
        s += v.x * v.x + v.y * v.y + v.z * v.z + v.w * v.w;
    }
    nrm[i] = s;
}
__global__ void pad_bias_kernel(const float* __restrict__ b, float* __restrict__ out) {
    out[threadIdx.x] = (threadIdx.x < 20) ? b[threadIdx.x] : 0.0f;
}
// LDS-tiled transpose-convert: x [32][1024][1024] (b,c,t) -> xT[(b*1024+t)*1024+c]
// 64x64 tiles; coalesced float4 reads, coalesced 8B bf16 writes.
__global__ __launch_bounds__(256)
void xpose_kernel(const float* __restrict__ x, unsigned short* __restrict__ xT) {
    __shared__ __align__(8) unsigned short tile[64][68];
    const int b  = blockIdx.z;
    const int c0 = blockIdx.y << 6;
    const int t0 = blockIdx.x << 6;
    const int tid = threadIdx.x;
#pragma unroll
    for (int k = 0; k < 4; ++k) {
        int ch = tid + (k << 8);              // 0..1023 float4 chunks
        int c = ch >> 4, tg = (ch & 15) << 2;
        float4 v = *(const float4*)(x + ((((long)b << 10) + c0 + c) << 10) + t0 + tg);
        tile[tg + 0][c] = f2bf(v.x);          // scatter transposed into LDS
        tile[tg + 1][c] = f2bf(v.y);
        tile[tg + 2][c] = f2bf(v.z);
        tile[tg + 3][c] = f2bf(v.w);
    }
    __syncthreads();
#pragma unroll
    for (int k = 0; k < 4; ++k) {
        int ch = tid + (k << 8);
        int t = ch >> 4, cg = (ch & 15) << 2;
        unsigned long long pk = *(const unsigned long long*)&tile[t][cg];
        *(unsigned long long*)(xT + ((((long)b << 10) + t0 + t) << 10) + c0 + cg) = pk;
    }
}
__global__ void angles_kernel(const float* __restrict__ dec, float* __restrict__ out) {
    long i = (long)blockIdx.x * blockDim.x + threadIdx.x;   // 32*20*1024
    int t = (int)(i & 1023); long bj = i >> 10; int j = (int)(bj % 20); long b = bj / 20;
    out[i] = dec[(((b << 10) + t) << 7) + j];
}
__global__ void indices_out_kernel(const int* __restrict__ idx, float* __restrict__ out) {
    long i = (long)blockIdx.x * blockDim.x + threadIdx.x;
    out[i] = (float)idx[i];
}
__global__ void scalars_kernel(const unsigned int* __restrict__ counts,
                               const float* __restrict__ lossAcc,
                               float* __restrict__ out)
{
    __shared__ float red[32];
    int tid = threadIdx.x;                         // 1024 threads
    float p = (float)counts[tid] * (1.0f / 131072.0f);
    float h = -p * logf(p + 1e-10f);
#pragma unroll
    for (int off = 16; off >= 1; off >>= 1) h += __shfl_xor(h, off, 32);
    if ((tid & 31) == 0) red[tid >> 5] = h;
    __syncthreads();
    if (tid < 32) {
        float v = red[tid];
#pragma unroll
        for (int off = 16; off >= 1; off >>= 1) v += __shfl_xor(v, off, 32);
        if (tid == 0) {
            float mse = lossAcc[0] * (1.0f / ((float)N_ROWS * (float)DIM));
            out[786432] = 1.25f * mse;   // vq_loss = cb + 0.25*commit (equal values)
            out[786433] = mse;           // codebook_loss
            out[786434] = mse;           // commit_loss
            out[786435] = expf(v);       // perplexity
        }
    }
}

// ---------------------------------------------------------------------------
extern "C" void kernel_launch(void* const* d_in, const int* in_sizes, int n_in,
                              void* d_out, int out_size, void* d_ws, size_t ws_size,
                              hipStream_t stream) {
    const float* x        = (const float*)d_in[0];
    const float* proj_W   = (const float*)d_in[1];
    const float* proj_b   = (const float*)d_in[2];
    const float* codebooks= (const float*)d_in[3];
    const float* dec_W1   = (const float*)d_in[4];
    const float* dec_b1   = (const float*)d_in[5];
    const float* dec_W2   = (const float*)d_in[6];
    const float* dec_b2   = (const float*)d_in[7];
    const float* dec_W3   = (const float*)d_in[8];
    const float* dec_b3   = (const float*)d_in[9];
    float* out = (float*)d_out;

    char* ws = (char*)d_ws;
    size_t off = 0;
    auto carve = [&](size_t bytes) -> char* {
        char* p = ws + off;
        off = (off + bytes + 255) & ~(size_t)255;
        return p;
    };
    unsigned short* xT    = (unsigned short*)carve((size_t)N_ROWS * INCH * 2);
    unsigned short* WpT   = (unsigned short*)carve((size_t)DIM * INCH * 2);
    unsigned short* W1T   = (unsigned short*)carve((size_t)1024 * DIM * 2);
    unsigned short* W2T   = (unsigned short*)carve((size_t)1024 * 1024 * 2);
    unsigned short* W3T   = (unsigned short*)carve((size_t)128 * 1024 * 2);
    float*          b3pad = (float*)carve(128 * 4);
    unsigned short* cbB   = (unsigned short*)carve((size_t)NLVL * NCODE * DIM * 2);
    float*          cbN   = (float*)carve((size_t)NLVL * NCODE * 4);
    float*          projF = (float*)carve((size_t)N_ROWS * DIM * 4);
    float*          residF= (float*)carve((size_t)N_ROWS * DIM * 4);
    unsigned short* residB= (unsigned short*)carve((size_t)N_ROWS * DIM * 2);
    unsigned short* qtotB = (unsigned short*)carve((size_t)N_ROWS * DIM * 2);
    unsigned short* h1    = (unsigned short*)carve((size_t)N_ROWS * 1024 * 2);
    unsigned short* h2    = (unsigned short*)carve((size_t)N_ROWS * 1024 * 2);
    float*          decF  = (float*)carve((size_t)N_ROWS * 128 * 4);
    int*            idxI  = (int*)carve((size_t)N_ROWS * NLVL * 4);
    unsigned int*   counts= (unsigned int*)carve(NCODE * 4);
    float*          lossA = (float*)carve(256);

    init_kernel<<<1, 1024, 0, stream>>>(counts, lossA);

    // weight / codebook preparation (transposed bf16 so GEMM tiles are contiguous-K)
    convert_transpose_kernel<<<(DIM * INCH + 255) / 256, 256, 0, stream>>>(proj_W, WpT, INCH, DIM, DIM);
    convert_transpose_kernel<<<(1024 * DIM + 255) / 256, 256, 0, stream>>>(dec_W1, W1T, DIM, 1024, 1024);
    convert_transpose_kernel<<<(1024 * 1024 + 255) / 256, 256, 0, stream>>>(dec_W2, W2T, 1024, 1024, 1024);
    convert_transpose_kernel<<<(128 * 1024 + 255) / 256, 256, 0, stream>>>(dec_W3, W3T, 1024, 20, 128);
    pad_bias_kernel<<<1, 128, 0, stream>>>(dec_b3, b3pad);
    cvt_kernel<<<(int)(((long)NLVL * NCODE * DIM + 255) / 256), 256, 0, stream>>>(
        codebooks, cbB, (long)NLVL * NCODE * DIM);
    cbnorm_kernel<<<16, 256, 0, stream>>>(codebooks, cbN);
    xpose_kernel<<<dim3(16, 16, 32), 256, 0, stream>>>(x, xT);

    // projection: proj = xflat @ Wp + b  -> f32 proj, f32 residual, bf16 residual
    gemm_bf16<0><<<dim3(DIM / 128, N_ROWS / 64), 256, 0, stream>>>(
        xT, WpT, proj_b, projF, residF, residB, INCH, DIM);

    // residual VQ: 4 levels of fused score/argmin + residual update
    for (int lvl = 0; lvl < NLVL; ++lvl) {
        rvq_score_kernel<<<N_ROWS / 32, 256, 0, stream>>>(
            residB, cbB + (size_t)lvl * NCODE * DIM, cbN + lvl * NCODE, idxI, counts, lvl);
        rvq_gather_kernel<<<(N_ROWS * 128) / 256, 256, 0, stream>>>(
            codebooks, idxI, residF, residB, lvl);
    }
    rvq_final_kernel<<<(N_ROWS * DIM / 4) / 256, 256, 0, stream>>>(projF, residF, qtotB, lossA);

    // decoder MLP
    gemm_bf16<1><<<dim3(1024 / 128, N_ROWS / 64), 256, 0, stream>>>(
        qtotB, W1T, dec_b1, nullptr, nullptr, h1, DIM, 1024);
    gemm_bf16<1><<<dim3(1024 / 128, N_ROWS / 64), 256, 0, stream>>>(
        h1, W2T, dec_b2, nullptr, nullptr, h2, 1024, 1024);
    gemm_bf16<2><<<dim3(1, N_ROWS / 64), 256, 0, stream>>>(
        h2, W3T, b3pad, decF, nullptr, nullptr, 1024, 128);

    // outputs: angles [32,20,1024], indices [32768,4] (as float), 4 scalars
    angles_kernel<<<(32 * 20 * 1024) / 256, 256, 0, stream>>>(decF, out);
    indices_out_kernel<<<(N_ROWS * NLVL) / 256, 256, 0, stream>>>(idxI, out + 655360);
    scalars_kernel<<<1, 1024, 0, stream>>>(counts, lossA, out);
}
